// CCGSupertaggerModel_32478542692817
// MI455X (gfx1250) — matrix operand order repaired and checked by reading, not compiled, not run on hardware
//
#include <hip/hip_runtime.h>
#include <hip/hip_bf16.h>

// ---------------------------------------------------------------------------
// CCG Supertagger (BERT-base fwd) for MI455X / gfx1250, wave32 + WMMA bf16.
// All GEMMs run through v_wmma_f32_16x16x32_bf16; f32 operands convert to
// bf16 (packed hw convert) while staging tiles into double-buffered LDS.
// Block tile 128x256, 8 waves of 64x64 -> 16 WMMA per 32-deep K step.
// ---------------------------------------------------------------------------

typedef __attribute__((ext_vector_type(16))) __bf16 v16bf;
typedef __attribute__((ext_vector_type(8)))  __bf16 v8bf;
typedef __attribute__((ext_vector_type(2)))  __bf16 v2bf;
typedef __attribute__((ext_vector_type(2)))  float  v2f;
typedef __attribute__((ext_vector_type(8)))  float  v8f;

// Packed f32x2 -> bf16x2 (lowered to the packed convert on gfx1250).
__device__ __forceinline__ unsigned int pack_bf16x2(float a, float b) {
  v2f f; f[0] = a; f[1] = b;
  union { v2bf v; unsigned int u; } c;
  c.v = __builtin_convertvector(f, v2bf);
  return c.u;
}

// ---------------------------------------------------------------------------
// Batched GEMM:  C[m,n] = act( sum_k A[m,k] * B(k,n) + bias[n] )
//   B(k,n) = TRANSB ? B[n*ldb + k] : B[k*ldb + n]
// Template params remove all runtime flag branching:
//   TRANSB  : B is row-major [N,K] (used for Q*K^T)
//   BOUNDS  : guard N edge (M % 128 == 0 and K % 32 == 0 always hold here)
//   PRERELU : relu(A) while converting to bf16
//   GELU    : exact gelu on the output
// Batch (grid.z): operand offset = (z/zdiv)*s1 + (z%zdiv)*s2.
// ---------------------------------------------------------------------------
template <int TRANSB, int BOUNDS, int PRERELU, int GELU>
__global__ __launch_bounds__(256)
void wmma_gemm_kernel(const float* __restrict__ A, const float* __restrict__ Bm,
                      const float* __restrict__ bias, float* __restrict__ Cc,
                      int M, int N, int K, int lda, int ldb, int ldc, int zdiv,
                      long long sA1, long long sA2, long long sB1, long long sB2,
                      long long sC1, long long sC2)
{
  __shared__ __attribute__((aligned(64))) unsigned short lA[2][128 * 32]; // [m][k]
  __shared__ __attribute__((aligned(64))) unsigned short lB[2][256 * 32]; // [n][k]

  const int z  = blockIdx.z;
  const int zq = z / zdiv, zr = z - zq * zdiv;
  const float* Ab = A  + zq * sA1 + zr * sA2;
  const float* Bb = Bm + zq * sB1 + zr * sB2;
  float*       Cb = Cc + zq * sC1 + zr * sC2;

  const int tid  = threadIdx.x;
  const int lane = tid & 31;
  const int wave = tid >> 5;
  const int wr   = wave >> 2;        // 0..1  (64-row slab)
  const int wc   = wave & 3;         // 0..3  (64-col slab)
  const int hi   = lane >> 4;        // lane half 0/1
  const int llo  = lane & 15;

  const int tileM = blockIdx.x * 128;
  const int tileN = blockIdx.y * 256;

  v8f acc[4][4];
#pragma unroll
  for (int mt = 0; mt < 4; ++mt)
#pragma unroll
    for (int nt = 0; nt < 4; ++nt)
#pragma unroll
      for (int r = 0; r < 8; ++r) acc[mt][nt][r] = 0.0f;

  // ---- tile staging (vectorized, branch-free in the fast path) ----
  auto stage = [&](int k0, int buf) {
    // A: 128x32, each thread 4 float4 chunks, packed 4xbf16 -> ds_store_b64
#pragma unroll
    for (int i = 0; i < 4; ++i) {
      int e  = tid + i * 256;
      int r  = e >> 3;
      int c4 = (e & 7) * 4;
      const float4 v =
          *reinterpret_cast<const float4*>(Ab + (long long)(tileM + r) * lda + k0 + c4);
      float a0 = v.x, a1 = v.y, a2 = v.z, a3 = v.w;
      if (PRERELU) {
        a0 = fmaxf(a0, 0.0f); a1 = fmaxf(a1, 0.0f);
        a2 = fmaxf(a2, 0.0f); a3 = fmaxf(a3, 0.0f);
      }
      uint2 p; p.x = pack_bf16x2(a0, a1); p.y = pack_bf16x2(a2, a3);
      *reinterpret_cast<uint2*>(&lA[buf][r * 32 + c4]) = p;
    }
    if (TRANSB) {
      // B row-major [N,K]: contiguous along k -> float4 loads
#pragma unroll
      for (int i = 0; i < 8; ++i) {
        int e  = tid + i * 256;
        int n  = e >> 3;
        int c4 = (e & 7) * 4;
        float b0 = 0.0f, b1 = 0.0f, b2 = 0.0f, b3 = 0.0f;
        if (!BOUNDS || (tileN + n) < N) {
          const float4 v =
              *reinterpret_cast<const float4*>(Bb + (long long)(tileN + n) * ldb + k0 + c4);
          b0 = v.x; b1 = v.y; b2 = v.z; b3 = v.w;
        }
        uint2 p; p.x = pack_bf16x2(b0, b1); p.y = pack_bf16x2(b2, b3);
        *reinterpret_cast<uint2*>(&lB[buf][n * 32 + c4]) = p;
      }
    } else {
      // B row-major [K,N]: column n = tid, 4 lane-coalesced b32 loads per chunk
      const int n = tid;
      const bool nok = !BOUNDS || (tileN + n) < N;
#pragma unroll
      for (int i = 0; i < 8; ++i) {
        const float* bp = Bb + (long long)(k0 + i * 4) * ldb + tileN + n;
        float b0 = 0.0f, b1 = 0.0f, b2 = 0.0f, b3 = 0.0f;
        if (nok) {
          b0 = bp[0]; b1 = bp[(long long)ldb];
          b2 = bp[2ll * ldb]; b3 = bp[3ll * ldb];
        }
        uint2 p; p.x = pack_bf16x2(b0, b1); p.y = pack_bf16x2(b2, b3);
        *reinterpret_cast<uint2*>(&lB[buf][n * 32 + i * 4]) = p;
      }
    }
  };

  const int nsteps = K >> 5;     // all K here are multiples of 32
  stage(0, 0);
  __syncthreads();

  for (int s = 0; s < nsteps; ++s) {
    const int cur = s & 1;

    // L2 prefetch two K-steps ahead (global_prefetch_b8 on gfx1250)
    if (s + 2 < nsteps) {
      int kp = (s + 2) << 5;
      __builtin_prefetch(Ab + (long long)(tileM + (tid >> 3)) * lda + kp + (tid & 7) * 4,
                         0, 1);
      if (TRANSB)
        __builtin_prefetch(Bb + (long long)(tileN + (tid >> 3)) * ldb + kp + (tid & 7) * 4,
                           0, 1);
      else
        __builtin_prefetch(Bb + (long long)kp * ldb + tileN + tid, 0, 1);
    }

    if (s + 1 < nsteps) stage((s + 1) << 5, cur ^ 1);   // overlap with compute

    const unsigned short* pA = lA[cur];
    const unsigned short* pB = lB[cur];

    // A fragments (ISA bf16 A layout: lanes0-15 K{0..7,16..23}, lanes16-31
    // K{8..15,24..31}; base = hi*8), B fragments use the same K layout.
    v16bf af[4];
#pragma unroll
    for (int mt = 0; mt < 4; ++mt) {
      int row = wr * 64 + mt * 16 + llo;
      union { v16bf v; v8bf h[2]; } fr;
      fr.h[0] = *reinterpret_cast<const v8bf*>(&pA[row * 32 + hi * 8]);
      fr.h[1] = *reinterpret_cast<const v8bf*>(&pA[row * 32 + hi * 8 + 16]);
      af[mt] = fr.v;
    }
    v16bf bfr[4];
#pragma unroll
    for (int nt = 0; nt < 4; ++nt) {
      int col = wc * 64 + nt * 16 + llo;
      union { v16bf v; v8bf h[2]; } fb;
      fb.h[0] = *reinterpret_cast<const v8bf*>(&pB[col * 32 + hi * 8]);
      fb.h[1] = *reinterpret_cast<const v8bf*>(&pB[col * 32 + hi * 8 + 16]);
      bfr[nt] = fb.v;
    }

#pragma unroll
    for (int mt = 0; mt < 4; ++mt)
#pragma unroll
      for (int nt = 0; nt < 4; ++nt)
        acc[mt][nt] = __builtin_amdgcn_wmma_f32_16x16x32_bf16(
            false, af[mt], false, bfr[nt], (short)0, acc[mt][nt], false, false);

    __syncthreads();
  }

  // ---- epilogue: C layout (VGPR r -> M = r + hi*8, N = llo) ----
#pragma unroll
  for (int nt = 0; nt < 4; ++nt) {
    const int n = tileN + wc * 64 + nt * 16 + llo;
    const bool nok = !BOUNDS || n < N;
    float bv = 0.0f;
    if (bias && nok) bv = bias[n];
#pragma unroll
    for (int mt = 0; mt < 4; ++mt) {
      const int mbase = tileM + wr * 64 + mt * 16 + hi * 8;
#pragma unroll
      for (int r = 0; r < 8; ++r) {
        if (nok) {
          float v = acc[mt][nt][r] + bv;
          if (GELU) v = 0.5f * v * (1.0f + erff(v * 0.70710678118654752f));
          Cb[(long long)(mbase + r) * ldc + n] = v;
        }
      }
    }
  }
}

// ---------------------------------------------------------------------------
// Embedding gather: x[b,s,:] = word_emb[tok] + pos_emb[s] + type_emb
// ---------------------------------------------------------------------------
__global__ __launch_bounds__(256)
void embed_kernel(const int* __restrict__ tok, const float* __restrict__ wemb,
                  const float* __restrict__ pemb, const float* __restrict__ temb,
                  float* __restrict__ x, int S, int H)
{
  int bs = blockIdx.x;
  int s  = bs % S;
  int t  = tok[bs];
  for (int h = threadIdx.x; h < H; h += 256)
    x[(long long)bs * H + h] =
        wemb[(long long)t * H + h] + pemb[(long long)s * H + h] + temb[h];
}

// ---------------------------------------------------------------------------
// out = LayerNorm(a (+ b)) * scale + bias, per row of length H (block/row)
// ---------------------------------------------------------------------------
__global__ __launch_bounds__(256)
void add_ln_kernel(float* __restrict__ out, const float* __restrict__ a,
                   const float* __restrict__ bsrc, const float* __restrict__ sc,
                   const float* __restrict__ bi, int H)
{
  __shared__ float red[256];
  __shared__ float red2[256];
  int row = blockIdx.x, tid = threadIdx.x;
  float loc[4];
  int nl = (H + 255) / 256;
  float s1 = 0.0f, s2 = 0.0f;
  for (int i = 0; i < nl; ++i) {
    int h = tid + i * 256;
    float v = 0.0f;
    if (h < H) {
      v = a[(long long)row * H + h];
      if (bsrc) v += bsrc[(long long)row * H + h];
    }
    loc[i] = v; s1 += v; s2 += v * v;
  }
  red[tid] = s1; red2[tid] = s2; __syncthreads();
  for (int o = 128; o > 0; o >>= 1) {
    if (tid < o) { red[tid] += red[tid + o]; red2[tid] += red2[tid + o]; }
    __syncthreads();
  }
  float mean = red[0] / (float)H;
  float var  = red2[0] / (float)H - mean * mean;
  float inv  = rsqrtf(var + 1e-12f);
  for (int i = 0; i < nl; ++i) {
    int h = tid + i * 256;
    if (h < H)
      out[(long long)row * H + h] = (loc[i] - mean) * inv * sc[h] + bi[h];
  }
}

// ---------------------------------------------------------------------------
// Row softmax over S=256 attention scores with scale + additive key mask.
// ---------------------------------------------------------------------------
__global__ __launch_bounds__(256)
void attn_softmax_kernel(float* __restrict__ att, const int* __restrict__ mask,
                         int S, int NH, float scale)
{
  __shared__ float red[256];
  int row = blockIdx.x, tid = threadIdx.x;
  int b = row / (NH * S);
  long long base = (long long)row * S;
  float v = att[base + tid] * scale +
            (1.0f - (float)mask[b * S + tid]) * -10000.0f;
  red[tid] = v; __syncthreads();
  for (int o = 128; o > 0; o >>= 1) {
    if (tid < o) red[tid] = fmaxf(red[tid], red[tid + o]);
    __syncthreads();
  }
  float mx = red[0]; __syncthreads();
  float e = expf(v - mx);
  red[tid] = e; __syncthreads();
  for (int o = 128; o > 0; o >>= 1) {
    if (tid < o) red[tid] += red[tid + o];
    __syncthreads();
  }
  att[base + tid] = e / red[0];
}

// ---------------------------------------------------------------------------
// Ragged word-piece mean pooling + class softmax. One block per (b, s_out).
// ---------------------------------------------------------------------------
__global__ __launch_bounds__(256)
void pool_softmax_kernel(const float* __restrict__ f2, const int* __restrict__ wpt,
                         float* __restrict__ out, int S, int W, int C)
{
  __shared__ float red[256];
  int bs = blockIdx.x;
  int b = bs / S, s = bs - b * S;
  int tid = threadIdx.x;

  int start = 0, cnt = 1;
  bool pooled = (s < W);
  if (pooled) {
    for (int j = 0; j < s; ++j) start += wpt[b * W + j];
    cnt = wpt[b * W + s];
  }

  float vals[2];
#pragma unroll
  for (int i = 0; i < 2; ++i) {
    int c = tid + i * 256;
    float v = -3.4e38f;
    if (c < C) {
      if (pooled) {
        float acc = 0.0f;
        for (int p = 0; p < cnt; ++p)
          acc += f2[(long long)(b * S + start + p) * C + c];
        v = acc / (float)cnt;
      } else {
        v = f2[(long long)bs * C + c];
      }
    }
    vals[i] = v;
  }
  float mx = fmaxf(vals[0], vals[1]);
  red[tid] = mx; __syncthreads();
  for (int o = 128; o > 0; o >>= 1) {
    if (tid < o) red[tid] = fmaxf(red[tid], red[tid + o]);
    __syncthreads();
  }
  mx = red[0]; __syncthreads();
  float e[2], lsum = 0.0f;
#pragma unroll
  for (int i = 0; i < 2; ++i) {
    int c = tid + i * 256;
    e[i] = (c < C) ? expf(vals[i] - mx) : 0.0f;
    lsum += e[i];
  }
  red[tid] = lsum; __syncthreads();
  for (int o = 128; o > 0; o >>= 1) {
    if (tid < o) red[tid] += red[tid + o];
    __syncthreads();
  }
  float inv = 1.0f / red[0];
#pragma unroll
  for (int i = 0; i < 2; ++i) {
    int c = tid + i * 256;
    if (c < C) out[(long long)bs * C + c] = e[i] * inv;
  }
}

// ---------------------------------------------------------------------------
// Host-side templated GEMM launcher
// ---------------------------------------------------------------------------
template <int TB, int BO, int PR, int GE>
static inline void run_gemm(hipStream_t stream,
                            const float* A, const float* Bm, const float* bias,
                            float* Cc, int M, int N, int K,
                            int lda, int ldb, int ldc, int Z, int zdiv,
                            long long sA1 = 0, long long sA2 = 0,
                            long long sB1 = 0, long long sB2 = 0,
                            long long sC1 = 0, long long sC2 = 0)
{
  dim3 g((M + 127) / 128, (N + 255) / 256, Z);
  wmma_gemm_kernel<TB, BO, PR, GE><<<g, 256, 0, stream>>>(
      A, Bm, bias, Cc, M, N, K, lda, ldb, ldc, zdiv,
      sA1, sA2, sB1, sB2, sC1, sC2);
}

// ---------------------------------------------------------------------------
// Host orchestration
// ---------------------------------------------------------------------------
extern "C" void kernel_launch(void* const* d_in, const int* in_sizes, int n_in,
                              void* d_out, int out_size, void* d_ws, size_t ws_size,
                              hipStream_t stream)
{
  (void)in_sizes; (void)n_in; (void)out_size; (void)ws_size;
  const int B = 16, S = 256, W = 128, H = 768, L = 12, NH = 12, DH = 64,
            FF = 3072, C = 425;
  const int T = B * S; // 4096 tokens

  const int*   tok   = (const int*)  d_in[0];
  const int*   mask  = (const int*)  d_in[1];
  const int*   wpt   = (const int*)  d_in[2];
  const float* wemb  = (const float*)d_in[3];
  const float* pemb  = (const float*)d_in[4];
  const float* temb  = (const float*)d_in[5];
  const float* elns  = (const float*)d_in[6];
  const float* elnb  = (const float*)d_in[7];
  const float* Wq    = (const float*)d_in[8];
  const float* bq    = (const float*)d_in[9];
  const float* Wk    = (const float*)d_in[10];
  const float* bk    = (const float*)d_in[11];
  const float* Wv    = (const float*)d_in[12];
  const float* bv    = (const float*)d_in[13];
  const float* Wo    = (const float*)d_in[14];
  const float* bo    = (const float*)d_in[15];
  const float* ln1s  = (const float*)d_in[16];
  const float* ln1b  = (const float*)d_in[17];
  const float* Wi    = (const float*)d_in[18];
  const float* bi    = (const float*)d_in[19];
  const float* Wo2   = (const float*)d_in[20];
  const float* bo2   = (const float*)d_in[21];
  const float* ln2s  = (const float*)d_in[22];
  const float* ln2b  = (const float*)d_in[23];
  const float* w1    = (const float*)d_in[24];
  const float* b1    = (const float*)d_in[25];
  const float* w2    = (const float*)d_in[26];
  const float* b2    = (const float*)d_in[27];

  // Workspace layout (floats). Peak ~120 MB.
  float* x   = (float*)d_ws;                 // [T,H]  hidden state
  float* q   = x   + (size_t)T * H;          // [T,H]
  float* kbf = q   + (size_t)T * H;          // [T,H]
  float* vbf = kbf + (size_t)T * H;          // [T,H]
  float* ctx = vbf + (size_t)T * H;          // [T,H]
  float* t0  = ctx + (size_t)T * H;          // max(B*NH*S*S, T*FF) = 12.58M f32
  float* f2  = t0  + (size_t)T * FF;         // [T,C]

  // ---- Embeddings + LN ----
  embed_kernel<<<T, 256, 0, stream>>>(tok, wemb, pemb, temb, x, S, H);
  add_ln_kernel<<<T, 256, 0, stream>>>(x, x, nullptr, elns, elnb, H);

  const long long sBH = (long long)S * H;     // per-batch stride in q/k/v/ctx
  const long long sSS = (long long)S * S;     // per-head score stride

  for (int l = 0; l < L; ++l) {
    const float* Wq_ = Wq + (size_t)l * H * H;   const float* bq_ = bq + (size_t)l * H;
    const float* Wk_ = Wk + (size_t)l * H * H;   const float* bk_ = bk + (size_t)l * H;
    const float* Wv_ = Wv + (size_t)l * H * H;   const float* bv_ = bv + (size_t)l * H;
    const float* Wo_ = Wo + (size_t)l * H * H;   const float* bo_ = bo + (size_t)l * H;
    const float* Wi_ = Wi + (size_t)l * H * FF;  const float* bi_ = bi + (size_t)l * FF;
    const float* W2_ = Wo2 + (size_t)l * FF * H; const float* b2_ = bo2 + (size_t)l * H;

    // QKV projections: [T,H] x [H,H]
    run_gemm<0,0,0,0>(stream, x, Wq_, bq_, q,   T, H, H, H, H, H, 1, 1);
    run_gemm<0,0,0,0>(stream, x, Wk_, bk_, kbf, T, H, H, H, H, H, 1, 1);
    run_gemm<0,0,0,0>(stream, x, Wv_, bv_, vbf, T, H, H, H, H, H, 1, 1);

    // scores[b,h] = Q_head (SxDH) * K_head^T, batched over z = b*NH + h
    run_gemm<1,0,0,0>(stream, q, kbf, nullptr, t0, S, S, DH, H, H, S,
                      B * NH, NH, sBH, DH, sBH, DH, (long long)NH * sSS, sSS);

    attn_softmax_kernel<<<B * NH * S, 256, 0, stream>>>(t0, mask, S, NH, 0.125f);

    // ctx[b,h] = att (SxS) * V_head (SxDH), written back interleaved [T,H]
    run_gemm<0,1,0,0>(stream, t0, vbf, nullptr, ctx, S, DH, S, S, H, H,
                      B * NH, NH, (long long)NH * sSS, sSS, sBH, DH, sBH, DH);

    // attention output projection + residual LN
    run_gemm<0,0,0,0>(stream, ctx, Wo_, bo_, q, T, H, H, H, H, H, 1, 1);
    add_ln_kernel<<<T, 256, 0, stream>>>(x, x, q,
                                         ln1s + (size_t)l * H, ln1b + (size_t)l * H, H);

    // FFN: gelu(x @ Wi + bi) @ Wo2 + bo2, residual LN
    run_gemm<0,0,0,1>(stream, x, Wi_, bi_, t0, T, FF, H, H, FF, FF, 1, 1);
    run_gemm<0,0,0,0>(stream, t0, W2_, b2_, q, T, H, FF, FF, H, H, 1, 1);
    add_ln_kernel<<<T, 256, 0, stream>>>(x, x, q,
                                         ln2s + (size_t)l * H, ln2b + (size_t)l * H, H);
  }

  // ---- MLP head: f1 = relu(x) @ w1 + b1 ; f2 = relu(f1) @ w2 + b2 ----
  run_gemm<0,0,1,0>(stream, x, w1, b1, t0, T, 1024, H, H, 1024, 1024, 1, 1);
  run_gemm<0,1,1,0>(stream, t0, w2, b2, f2, T, C, 1024, 1024, C, C, 1, 1);

  // ---- ragged mean pooling + class softmax ----
  pool_softmax_kernel<<<T, 256, 0, stream>>>(f2, wpt, (float*)d_out, S, W, C);
}
